// MultiHeadSelfAttention_62637803045169
// MI455X (gfx1250) — compile-verified, matrix operand order checked by reading, not run on hardware
//
#include <hip/hip_runtime.h>
#include <hip/hip_fp16.h>

// ---------------------------------------------------------------------------
// MHA forward for MI455X (gfx1250, wave32). All matmuls via
// v_wmma_f32_16x16x32_f16; global->LDS staging via async-to-LDS DMA
// (ASYNCcnt) with double buffering.
// ---------------------------------------------------------------------------

typedef _Float16 v16h __attribute__((ext_vector_type(16)));
typedef _Float16 v8h  __attribute__((ext_vector_type(8)));
typedef float    v8f  __attribute__((ext_vector_type(8)));
typedef int      v4i  __attribute__((ext_vector_type(4)));

typedef __attribute__((address_space(1))) v4i as1_v4i;
typedef __attribute__((address_space(3))) v4i as3_v4i;

#define HIDDEN 1024
#define HEADS  16
#define HD     64
#define BATCH  4
#define SEQ    2048
#define MTOT   (BATCH * SEQ)   // 8192 rows

// ---------------------------------------------------------------------------
// Async 16-byte global->LDS copy (per lane), tracked by ASYNCcnt.
// ---------------------------------------------------------------------------
__device__ __forceinline__ void async_copy16(const _Float16* g, _Float16* l) {
#if __has_builtin(__builtin_amdgcn_global_load_async_to_lds_b128)
  __builtin_amdgcn_global_load_async_to_lds_b128((as1_v4i*)g, (as3_v4i*)l, 0, 0);
#else
  *(v4i*)l = *(const v4i*)g;   // synchronous fallback
#endif
}

__device__ __forceinline__ void wait_async() {
#if __has_builtin(__builtin_amdgcn_s_wait_asynccnt)
  __builtin_amdgcn_s_wait_asynccnt(0);
#else
  asm volatile("s_wait_asynccnt 0x0" ::: "memory");
#endif
  asm volatile("" ::: "memory");   // compiler fence: LDS now holds new data
}

// ---------------------------------------------------------------------------
// f32 -> f16 conversion (grid-stride)
// ---------------------------------------------------------------------------
__global__ void f32_to_f16_kernel(const float* __restrict__ src,
                                  _Float16* __restrict__ dst, int n) {
  int i = blockIdx.x * blockDim.x + threadIdx.x;
  int stride = gridDim.x * blockDim.x;
  for (; i < n; i += stride) dst[i] = (_Float16)src[i];
}

// ---------------------------------------------------------------------------
// Fragment loader: 16-half A/B fragment for one lane.
// Lanes 0-15 take K halves {khalf..+7, 16+khalf..+7}; khalf = 8*(lane>=16).
// rowptr must be 16-byte aligned at both chunk offsets.
// ---------------------------------------------------------------------------
__device__ __forceinline__ v16h load_frag(const _Float16* rowptr, int khalf) {
  union { v16h v; v8h h[2]; } u;
  u.h[0] = *(const v8h*)(rowptr + khalf);
  u.h[1] = *(const v8h*)(rowptr + 16 + khalf);
  return u.v;
}

// ---------------------------------------------------------------------------
// GEMM: C(8192 x 1024) = A(8192x1024) * W^T + bias, W (N x K) row-major.
// mode 0/1/2: Q/K/V -> f16 scattered to (B,H,T,hd); mode 0 folds 0.125 scale.
// mode 3:     O-projection -> f32 row-major output.
// Block: 256 threads = 8 waves; tile BM=128 x BN=64 x BK=32, double-buffered
// async LDS staging. Wave w -> 32x32 sub-tile = 2x2 WMMA tiles.
// ---------------------------------------------------------------------------
#define BM 128
#define BN 64
#define BK 32
#define LDT (BK + 8)   // LDS row stride in halfs (80 B, 16-B aligned)

__launch_bounds__(256)
__global__ void gemm_wmma_kernel(const _Float16* __restrict__ A,
                                 const _Float16* __restrict__ W,
                                 const float* __restrict__ bias,
                                 void* __restrict__ out, int mode) {
  __shared__ _Float16 As[2][BM * LDT];
  __shared__ _Float16 Ws[2][BN * LDT];

  const int tid   = threadIdx.x;
  const int lane  = tid & 31;
  const int wave  = tid >> 5;
  const int waveM = wave & 3;      // 0..3
  const int waveN = wave >> 2;     // 0..1
  const int Mbase = blockIdx.y * BM;
  const int Nbase = blockIdx.x * BN;

  const int row16 = lane & 15;
  const int khalf = (lane >> 4) << 3;  // 0 or 8 halfs

  // fixed staging coordinates: A chunks tid and tid+256, W chunk tid
  const int ra0 = tid >> 2,           ca0 = (tid & 3) << 3;
  const int ra1 = (tid + 256) >> 2,   ca1 = (tid & 3) << 3;
  const int rw  = tid >> 2,           cw  = (tid & 3) << 3;

  auto issue_tiles = [&](int buf, int k0) {
    async_copy16(&A[(size_t)(Mbase + ra0) * HIDDEN + k0 + ca0], &As[buf][ra0 * LDT + ca0]);
    async_copy16(&A[(size_t)(Mbase + ra1) * HIDDEN + k0 + ca1], &As[buf][ra1 * LDT + ca1]);
    async_copy16(&W[(size_t)(Nbase + rw)  * HIDDEN + k0 + cw],  &Ws[buf][rw  * LDT + cw]);
  };

  v8f c00 = {}, c01 = {}, c10 = {}, c11 = {};

  const int nk = HIDDEN / BK;
  issue_tiles(0, 0);

  for (int kt = 0; kt < nk; ++kt) {
    const int cur = kt & 1;
    wait_async();            // this wave's tile-kt copies landed in LDS
    __syncthreads();         // ... and everyone else's
    if (kt + 1 < nk) issue_tiles(cur ^ 1, (kt + 1) * BK);   // overlap next tile
    if (kt + 2 < nk)
      __builtin_prefetch(&A[(size_t)(Mbase + (tid >> 1)) * HIDDEN + (kt + 2) * BK], 0, 3);

    const _Float16* aRow0 = &As[cur][(waveM * 32 + row16) * LDT];
    const _Float16* aRow1 = &As[cur][(waveM * 32 + 16 + row16) * LDT];
    const _Float16* wRow0 = &Ws[cur][(waveN * 32 + row16) * LDT];
    const _Float16* wRow1 = &Ws[cur][(waveN * 32 + 16 + row16) * LDT];
    v16h a0 = load_frag(aRow0, khalf);
    v16h a1 = load_frag(aRow1, khalf);
    v16h b0 = load_frag(wRow0, khalf);
    v16h b1 = load_frag(wRow1, khalf);

    c00 = __builtin_amdgcn_wmma_f32_16x16x32_f16(false, a0, false, b0, (short)0, c00, false, false);
    c01 = __builtin_amdgcn_wmma_f32_16x16x32_f16(false, a0, false, b1, (short)0, c01, false, false);
    c10 = __builtin_amdgcn_wmma_f32_16x16x32_f16(false, a1, false, b0, (short)0, c10, false, false);
    c11 = __builtin_amdgcn_wmma_f32_16x16x32_f16(false, a1, false, b1, (short)0, c11, false, false);
    __syncthreads();         // all waves done reading buffer `cur`
  }

  // C layout: N = lane&15; M = r + 8*(lane>=16)
  const int Ml = (lane >> 4) << 3;
  const int Nl = lane & 15;
  float*    outF = (float*)out;
  _Float16* outH = (_Float16*)out;

#pragma unroll
  for (int mt = 0; mt < 2; ++mt) {
#pragma unroll
    for (int nt = 0; nt < 2; ++nt) {
      v8f c = (mt == 0) ? ((nt == 0) ? c00 : c01) : ((nt == 0) ? c10 : c11);
#pragma unroll
      for (int r = 0; r < 8; ++r) {
        int grow = Mbase + waveM * 32 + mt * 16 + Ml + r;
        int gcol = Nbase + waveN * 32 + nt * 16 + Nl;
        float v = c[r] + bias[gcol];
        if (mode == 3) {
          outF[(size_t)grow * HIDDEN + gcol] = v;
        } else {
          if (mode == 0) v *= 0.125f;  // fold 1/sqrt(hd) into Q
          int b = grow >> 11, t = grow & (SEQ - 1);
          int h = gcol >> 6,  d = gcol & 63;
          outH[(((size_t)(b * HEADS + h) * SEQ + t) << 6) + d] = (_Float16)v;
        }
      }
    }
  }
}

// ---------------------------------------------------------------------------
// Flash attention: one wave per 16-query tile of one (b,h); 32-key k-tiles
// double-buffered in wave-private LDS via async-to-LDS DMA (no barriers --
// ASYNCcnt wait alone synchronizes a wave's own slices). Online softmax in
// C-layout (rows confined to a 16-lane half -> shfl_xor 1/2/4/8); P goes
// through LDS to A-layout; PV = 4 WMMAs per k-tile.
// ---------------------------------------------------------------------------
#define KT  32            // keys per tile
#define LDK (HD + 8)      // 72 halfs = 144 B row stride (16-B aligned)
#define LDP (KT + 8)      // 40 halfs = 80 B row stride (16-B aligned)

__launch_bounds__(128)
__global__ void attn_wmma_kernel(const _Float16* __restrict__ Qh,
                                 const _Float16* __restrict__ Kh,
                                 const _Float16* __restrict__ Vh,
                                 const unsigned char* __restrict__ amask,
                                 _Float16* __restrict__ ctx) {
  __shared__ _Float16 Ks[4][2][KT * LDK];
  __shared__ _Float16 Vs[4][2][KT * LDK];
  __shared__ _Float16 Ps[4][16 * LDP];

  const int lane = threadIdx.x & 31;
  const int wave = threadIdx.x >> 5;
  const int g    = blockIdx.x * 4 + wave;      // 0 .. B*H*(T/16)-1 = 8191
  const int qbase = (g & 127) << 4;            // 128 q-tiles per (b,h)
  const int bh = g >> 7;
  const int b = bh >> 4;
  const int h = bh & 15;

  const int row16 = lane & 15;
  const int khalf = (lane >> 4) << 3;
  const int Ml = (lane >> 4) << 3;
  const int Nl = lane & 15;

  _Float16* PsW = Ps[wave];
  const size_t headBase = ((size_t)(b * HEADS + h) * SEQ);

  auto issue_kv = [&](int buf, int jbase) {
    const _Float16* ksrc = Kh + ((headBase + jbase + lane) << 6);
    const _Float16* vsrc = Vh + ((headBase + jbase + lane) << 6);
#pragma unroll
    for (int cc = 0; cc < HD; cc += 8) {
      async_copy16(ksrc + cc, &Ks[wave][buf][lane * LDK + cc]);
      async_copy16(vsrc + cc, &Vs[wave][buf][lane * LDK + cc]);
    }
  };

  // Q tile (16 x 64) as two A fragments (0.125 scale already folded in).
  const _Float16* qrow = Qh + ((headBase + qbase + row16) << 6);
  v16h qa0 = load_frag(qrow, khalf);
  v16h qa1 = load_frag(qrow + 32, khalf);

  v8f o[4] = {v8f{}, v8f{}, v8f{}, v8f{}};
  float mrun[8], lrun[8];
#pragma unroll
  for (int r = 0; r < 8; ++r) { mrun[r] = -1e30f; lrun[r] = 0.0f; }

  const int ntiles = (qbase + 47) >> 5;   // cover keys 0 .. qbase+15
  issue_kv(0, 0);

  for (int jt = 0; jt < ntiles; ++jt) {
    const int cur = jt & 1;
    const int jbase = jt * KT;
    wait_async();                                     // tile jt resident
    if (jt + 1 < ntiles) issue_kv(cur ^ 1, jbase + KT);  // overlap next tile

    const _Float16* KsW = Ks[wave][cur];
    const _Float16* VsW = Vs[wave][cur];

    // ---- S = Q K^T : two 16x16 C-frags, each 2 chained WMMAs over d ----
    v8f s[2];
#pragma unroll
    for (int sh = 0; sh < 2; ++sh) {
      const _Float16* krow = &KsW[(sh * 16 + row16) * LDK];
      v16h kb0 = load_frag(krow, khalf);        // d 0..31
      v16h kb1 = load_frag(krow + 32, khalf);   // d 32..63
      v8f acc = {};
      acc = __builtin_amdgcn_wmma_f32_16x16x32_f16(false, qa0, false, kb0, (short)0, acc, false, false);
      acc = __builtin_amdgcn_wmma_f32_16x16x32_f16(false, qa1, false, kb1, (short)0, acc, false, false);
      s[sh] = acc;
    }

    // ---- causal + padding mask (branchless) ----
    const unsigned char* mptr = amask + (size_t)b * SEQ + jbase;
#pragma unroll
    for (int sh = 0; sh < 2; ++sh) {
      int j = jbase + sh * 16 + Nl;
      bool colok = (mptr[sh * 16 + Nl] != 0);
#pragma unroll
      for (int r = 0; r < 8; ++r) {
        int t = qbase + Ml + r;
        bool ok = colok && (j <= t);
        s[sh][r] = ok ? s[sh][r] : -1e30f;
      }
    }

    // ---- online softmax (rows confined to a 16-lane half) ----
    float alphas[8];
#pragma unroll
    for (int r = 0; r < 8; ++r) {
      float mx = fmaxf(s[0][r], s[1][r]);
#pragma unroll
      for (int off = 8; off >= 1; off >>= 1)
        mx = fmaxf(mx, __shfl_xor(mx, off, 32));
      float mnew  = fmaxf(mrun[r], mx);
      float alpha = __expf(mrun[r] - mnew);
      float p0 = __expf(s[0][r] - mnew);
      float p1 = __expf(s[1][r] - mnew);
      float rs = p0 + p1;
#pragma unroll
      for (int off = 8; off >= 1; off >>= 1)
        rs += __shfl_xor(rs, off, 32);
      lrun[r] = lrun[r] * alpha + rs;
      mrun[r] = mnew;
      alphas[r] = alpha;
      s[0][r] = p0;
      s[1][r] = p1;
    }

    // ---- rescale O; write P to LDS (C-layout -> row-major transpose) ----
#pragma unroll
    for (int r = 0; r < 8; ++r) {
      o[0][r] *= alphas[r]; o[1][r] *= alphas[r];
      o[2][r] *= alphas[r]; o[3][r] *= alphas[r];
      PsW[(Ml + r) * LDP + Nl]      = (_Float16)s[0][r];
      PsW[(Ml + r) * LDP + 16 + Nl] = (_Float16)s[1][r];
    }

    // ---- P as A fragment (16x32); DS ops are in-order within the wave ----
    v16h pa = load_frag(&PsW[row16 * LDP], khalf);

    // ---- O += P V : 4 WMMAs (V B-frags gathered from LDS, keys in halfs) --
#pragma unroll
    for (int nt = 0; nt < 4; ++nt) {
      v16h vb = {};
#pragma unroll
      for (int vh = 0; vh < 8; ++vh) {
        int k0 = (vh < 4) ? (2 * vh) : (2 * vh + 8);   // base K mapping
        int kk = k0 + khalf;                            // +8 for upper half-wave
        vb[2 * vh]     = VsW[kk * LDK + nt * 16 + Nl];
        vb[2 * vh + 1] = VsW[(kk + 1) * LDK + nt * 16 + Nl];
      }
      o[nt] = __builtin_amdgcn_wmma_f32_16x16x32_f16(false, pa, false, vb, (short)0, o[nt], false, false);
    }
  }

  // ---- epilogue: ctx[(b*T+t), h*64+d] = O / l ----
#pragma unroll
  for (int nt = 0; nt < 4; ++nt) {
#pragma unroll
    for (int r = 0; r < 8; ++r) {
      float v = o[nt][r] / lrun[r];
      int t = qbase + Ml + r;
      int col = h * HD + nt * 16 + Nl;
      ctx[((size_t)b * SEQ + t) * HIDDEN + col] = (_Float16)v;
    }
  }
}

// ---------------------------------------------------------------------------
// Host launcher
// ---------------------------------------------------------------------------
extern "C" void kernel_launch(void* const* d_in, const int* in_sizes, int n_in,
                              void* d_out, int out_size, void* d_ws, size_t ws_size,
                              hipStream_t stream) {
  const float* x   = (const float*)d_in[0];
  const unsigned char* amask = (const unsigned char*)d_in[1];
  const float* q_w = (const float*)d_in[2];
  const float* q_b = (const float*)d_in[3];
  const float* k_w = (const float*)d_in[4];
  const float* k_b = (const float*)d_in[5];
  const float* v_w = (const float*)d_in[6];
  const float* v_b = (const float*)d_in[7];
  const float* o_w = (const float*)d_in[8];
  const float* o_b = (const float*)d_in[9];
  float* out = (float*)d_out;

  // Workspace layout (f16 elements)
  _Float16* ws = (_Float16*)d_ws;
  size_t off = 0;
  const size_t NX = (size_t)MTOT * HIDDEN;     // 8.39M
  const size_t NW = (size_t)HIDDEN * HIDDEN;   // 1.05M
  _Float16* xh   = ws + off; off += NX;
  _Float16* wqh  = ws + off; off += NW;
  _Float16* wkh  = ws + off; off += NW;
  _Float16* wvh  = ws + off; off += NW;
  _Float16* woh  = ws + off; off += NW;
  _Float16* Qh   = ws + off; off += NX;
  _Float16* Kh   = ws + off; off += NX;
  _Float16* Vh   = ws + off; off += NX;
  _Float16* ctxh = ws + off; off += NX;

  // 1) f32 -> f16 conversions
  f32_to_f16_kernel<<<2048, 256, 0, stream>>>(x,   xh,  (int)NX);
  f32_to_f16_kernel<<<512,  256, 0, stream>>>(q_w, wqh, (int)NW);
  f32_to_f16_kernel<<<512,  256, 0, stream>>>(k_w, wkh, (int)NW);
  f32_to_f16_kernel<<<512,  256, 0, stream>>>(v_w, wvh, (int)NW);
  f32_to_f16_kernel<<<512,  256, 0, stream>>>(o_w, woh, (int)NW);

  // 2) Q/K/V projections (WMMA GEMM, epilogue scatters to (B,H,T,hd))
  dim3 ggrid(HIDDEN / BN, MTOT / BM);  // (16, 64)
  gemm_wmma_kernel<<<ggrid, 256, 0, stream>>>(xh, wqh, q_b, (void*)Qh, 0);
  gemm_wmma_kernel<<<ggrid, 256, 0, stream>>>(xh, wkh, k_b, (void*)Kh, 1);
  gemm_wmma_kernel<<<ggrid, 256, 0, stream>>>(xh, wvh, v_b, (void*)Vh, 2);

  // 3) Flash attention: B*H*(T/16) = 8192 wave-tiles, 4 waves/block
  attn_wmma_kernel<<<2048, 128, 0, stream>>>(Qh, Kh, Vh, amask, ctxh);

  // 4) Output projection -> f32 d_out
  gemm_wmma_kernel<<<ggrid, 256, 0, stream>>>(ctxh, woh, o_b, (void*)out, 3);
}
